// PlanningHorizon_42322607734941
// MI455X (gfx1250) — matrix operand order repaired and checked by reading, not compiled, not run on hardware
//
#include <hip/hip_runtime.h>
#include <hip/hip_bf16.h>

typedef _Float16 v8h  __attribute__((ext_vector_type(8)));
typedef _Float16 v16h __attribute__((ext_vector_type(16)));
typedef float    v8f  __attribute__((ext_vector_type(8)));

namespace {
constexpr int Cc = 20, Bb = 256, Hh = 10, Ss = 64, Aa = 16, Hd = 128, Rh = 64;
constexpr int CB   = Cc * Bb;   // 5120 rollout rows
constexpr int ROWS = 16;        // rows per workgroup (one WMMA M tile)
constexpr int SNN  = 15;
constexpr float BETA = 0.9f, THR = 1.0f;

// packed f16 weight offsets (in halves) inside d_ws
constexpr int OFF_WSE = 0;        // [64 x128] K=64   -> 8 nt * 2 ks * 512 = 8192
constexpr int OFF_WAE = 8192;     // [16 x128] Kpad32 -> 8 * 1 * 512      = 4096
constexpr int OFF_W1  = 12288;    // [256x128] K=256  -> 8 * 8 * 512      = 32768
constexpr int OFF_W2  = 45056;    // [128x128] K=128  -> 8 * 4 * 512      = 16384
constexpr int OFF_WDE = 61440;    // [128x 64] K=128  -> 4 * 4 * 512      = 8192
constexpr int OFF_WUN = 69632;    // [128x 64] K=128  -> 4 * 4 * 512      = 8192
constexpr int OFF_WR1 = 77824;    // [ 80x 64] Kpad96 -> 4 * 3 * 512      = 6144
constexpr int OFF_WR2 = 83968;    // [ 64x 64] K=64   -> 4 * 2 * 512      = 4096
constexpr int PACK_TOTAL = 88064; // halves
constexpr size_t CUM_BYTE_OFF = (size_t)PACK_TOTAL * sizeof(_Float16); // 176128 B
} // namespace

// ---------------------------------------------------------------------------
// Pack one f32 weight matrix W[K][N] (row-major, x @ W convention) into the
// WMMA B-operand fragment layout:
//   idx = ((nt*ksteps + ks)*32 + lane)*16 + j
//   k   = ks*32 + (lane>=16 ? 16 : 0) + j   (zero-padded past K)
//   n   = nt*16 + (lane & 15)
// ---------------------------------------------------------------------------
__global__ void pack_w_kernel(const float* __restrict__ W, _Float16* __restrict__ dst,
                              int K, int N, int Kpad) {
    int ksteps = Kpad >> 5;
    int total  = (N >> 4) * ksteps * 512;
    int idx = blockIdx.x * blockDim.x + threadIdx.x;
    if (idx >= total) return;
    int j    = idx & 15;
    int lane = (idx >> 4) & 31;
    int ks   = (idx >> 9) % ksteps;
    int nt   = idx / (ksteps << 9);
    int k    = ks * 32 + ((lane >> 4) << 4) + j;
    int n    = nt * 16 + (lane & 15);
    float v  = (k < K) ? W[(size_t)k * N + n] : 0.0f;
    dst[idx] = (_Float16)v;
}

// A fragment (16x32 f16) from an LDS row-major [16][stride] buffer.
// lane<16: row=lane,  K = k0+0..15
// lane>=16: row=lane-16, K = k0+16..31
__device__ __forceinline__ v16h ldsA(const _Float16* base, int stride, int lane, int k0) {
    const _Float16* p = base + (lane & 15) * stride + k0 + ((lane >> 4) << 4);
    v8h lo = *(const v8h*)p;
    v8h hi = *(const v8h*)(p + 8);
    return __builtin_shufflevector(lo, hi, 0, 1, 2, 3, 4, 5, 6, 7,
                                   8, 9, 10, 11, 12, 13, 14, 15);
}

// One 16x16 f32 output tile: A (LDS activations) x B (packed global weights).
template <int KSTEPS>
__device__ __forceinline__ v8f gemm_tile(const _Float16* A, int strideA,
                                         const _Float16* __restrict__ Bp,
                                         int nt, int lane) {
    const _Float16* bbase = Bp + ((size_t)nt * KSTEPS * 512) + lane * 16;
    v8f acc = {};
#pragma unroll
    for (int ks = 0; ks < KSTEPS; ++ks) {
        v16h a = ldsA(A, strideA, lane, ks * 32);
        v16h b = *(const v16h*)(bbase + ks * 512);
        acc = __builtin_amdgcn_wmma_f32_16x16x32_f16(false, a, false, b,
                                                     (short)0, acc, false, false);
    }
    return acc;
}

// Scatter a 16x16 f32 accumulator tile to LDS as f16 (row-major [16][stride]).
__device__ __forceinline__ void storeD(_Float16* dst, int stride, int n0, int lane,
                                       const v8f& c) {
    int col = n0 + (lane & 15);
    int rb  = (lane >> 4) << 3;
#pragma unroll
    for (int e = 0; e < 8; ++e) dst[(rb + e) * stride + col] = (_Float16)c[e];
}

__device__ __forceinline__ float softplusf(float x) {
    return fmaxf(x, 0.0f) + log1pf(expf(-fabsf(x)));
}

// ---------------------------------------------------------------------------
// Main rollout: each workgroup owns 16 rows (g = g0..g0+15, g = c*B + b).
// 8 waves; wave w owns Hd column tile w (one 16x16 WMMA tile -> low VGPR
// pressure, no extended-VGPR-bank traffic). Decoder splits mean (waves 0-3)
// and var (waves 4-7) across waves, communicating sd through LDS.
// ---------------------------------------------------------------------------
__global__ __launch_bounds__(256)
void rollout_kernel(const float* __restrict__ state0,   // [B,S]
                    const float* __restrict__ actions,  // [C,B,H,A]
                    const float* __restrict__ noise,    // [C,B,H,S]
                    const _Float16* __restrict__ wsP,   // packed weights
                    const float* __restrict__ bse, const float* __restrict__ bae,
                    const float* __restrict__ b1,  const float* __restrict__ b2,
                    const float* __restrict__ bdec, const float* __restrict__ bunc,
                    const float* __restrict__ br1, const float* __restrict__ br2,
                    const float* __restrict__ br3, const float* __restrict__ Wr3,
                    float* __restrict__ cumOut) {
    __shared__ __align__(16) _Float16 stateB[ROWS * 72];     // S=64 (+pad)
    __shared__ __align__(16) _Float16 actB[ROWS * 40];       // A=16 padded to K=32
    __shared__ __align__(16) _Float16 catB[ROWS * 264];      // concat(se,ae) K=256
    __shared__ __align__(16) _Float16 spkB[2 * ROWS * 136];  // double-buffered spikes
    __shared__ __align__(16) _Float16 mem2B[ROWS * 136];
    __shared__ __align__(16) _Float16 sdB[ROWS * 72];        // std-dev handoff
    __shared__ __align__(16) _Float16 xB[ROWS * 104];        // [state|act] K=80 pad 96
    __shared__ __align__(16) _Float16 h1B[ROWS * 72];
    __shared__ __align__(16) _Float16 h2B[ROWS * 72];

    const int tid  = threadIdx.x;
    const int w    = tid >> 5;     // wave 0..7
    const int lane = tid & 31;
    const int g0   = blockIdx.x * ROWS;

    const _Float16* WseP = wsP + OFF_WSE;
    const _Float16* WaeP = wsP + OFF_WAE;
    const _Float16* W1P  = wsP + OFF_W1;
    const _Float16* W2P  = wsP + OFF_W2;
    const _Float16* WdeP = wsP + OFF_WDE;
    const _Float16* WunP = wsP + OFF_WUN;
    const _Float16* Wr1P = wsP + OFF_WR1;
    const _Float16* Wr2P = wsP + OFF_WR2;

    // --- init LDS: zero K-padding regions once, load initial state ---
    for (int i = tid; i < ROWS * 40; i += 256) actB[i] = (_Float16)0.0f;
    for (int i = tid; i < ROWS * 104; i += 256) xB[i] = (_Float16)0.0f;
    for (int i = tid; i < ROWS * Ss; i += 256) {
        int r = i >> 6, s = i & 63;
        stateB[r * 72 + s] = (_Float16)state0[(size_t)((g0 + r) & (Bb - 1)) * Ss + s];
    }

    const int colH = lane & 15;
    const int rb   = (lane >> 4) << 3;
    const int n0h  = w * 16;                 // this wave's Hd tile
    const int ws4  = w & 3;
    const int n0s  = ws4 * 16;               // S/Rh tile for split phases
    const float bsev = bse[n0h + colH], baev = bae[n0h + colH];
    const float b1v  = b1[n0h + colH],  b2v  = b2[n0h + colH];
    const float bdecv = bdec[n0s + colH], buncv = bunc[n0s + colH];
    const float br1v = br1[n0s + colH], br2v = br2[n0s + colH], br3v = br3[0];
    float cumr = 0.0f;
    __syncthreads();

#pragma unroll 1
    for (int h = 0; h < Hh; ++h) {
        // (a) stage actions; prefetch this step's noise rows into cache
        if (tid < ROWS * Aa) {
            int r = tid >> 4, a = tid & 15;
            float av = actions[(size_t)(g0 + r) * (Hh * Aa) + h * Aa + a];
            actB[r * 40 + a]     = (_Float16)av;
            xB[r * 104 + Ss + a] = (_Float16)av;
        } else if (tid < ROWS * Aa + 32) {
            int p = tid - ROWS * Aa;          // 32 threads: 16 rows x 2 halves
            int r = p >> 1, half = (p & 1) * 32;
            __builtin_prefetch(&noise[(size_t)(g0 + r) * (Hh * Ss) + h * Ss + half], 0, 3);
        }
        __syncthreads();

        // (b) encoders -> concat buffer (wave w: Hd tile w)
        {
            v8f se = gemm_tile<2>(stateB, 72, WseP, w, lane);
            v8f ae = gemm_tile<1>(actB, 40, WaeP, w, lane);
#pragma unroll
            for (int e = 0; e < 8; ++e) {
                se[e] = fmaxf(se[e] + bsev, 0.0f);
                ae[e] = fmaxf(ae[e] + baev, 0.0f);
            }
            storeD(catB, 264, n0h, lane, se);
            storeD(catB, 264, Hd + n0h, lane, ae);
        }
        __syncthreads();

        // (c) cur = concat @ W1 + b1 (constant over SNN steps, lives in regs)
        v8f cur = gemm_tile<8>(catB, 264, W1P, w, lane);
#pragma unroll
        for (int e = 0; e < 8; ++e) cur[e] += b1v;

        // (d) SNN leaky integrate-and-fire, 15 steps, membranes in registers
        v8f m1 = {}, m2 = {};
        int buf = 0;
        const int spkCol = n0h + colH;
#pragma unroll 1
        for (int t = 0; t < SNN; ++t) {
            _Float16* spkDst = spkB + buf * (ROWS * 136);
#pragma unroll
            for (int e = 0; e < 8; ++e) {
                float so = (m1[e] > THR) ? 1.0f : 0.0f;   // reset uses prev spike
                float nm = BETA * m1[e] + cur[e] - so;     // THR*spike, THR=1
                m1[e] = nm;
                // spikes are exactly 0/1: select f16 constants, no cvt chain
                spkDst[(rb + e) * 136 + spkCol] =
                    (nm > THR) ? (_Float16)1.0f : (_Float16)0.0f;
            }
            __syncthreads();
            v8f c2 = gemm_tile<4>(spkDst, 136, W2P, w, lane);
#pragma unroll
            for (int e = 0; e < 8; ++e) {
                float so = (m2[e] > THR) ? 1.0f : 0.0f;
                m2[e] = BETA * m2[e] + (c2[e] + b2v) - so;
            }
            buf ^= 1;
        }

        // (e) share mem2 for decoders
        storeD(mem2B, 136, n0h, lane, m2);
        __syncthreads();

        // (f) decode: waves 4-7 produce sd tiles, waves 0-3 produce mean tiles
        v8f mean = {};
        if (w >= 4) {
            v8f varr = gemm_tile<4>(mem2B, 136, WunP, ws4, lane);
#pragma unroll
            for (int e = 0; e < 8; ++e) {
                float sd = sqrtf(softplusf(varr[e] + buncv) + 1e-8f);
                sdB[(rb + e) * 72 + n0s + colH] = (_Float16)sd;
            }
        } else {
            mean = gemm_tile<4>(mem2B, 136, WdeP, ws4, lane);
        }
        __syncthreads();
        if (w < 4) {   // sample next state
#pragma unroll
            for (int e = 0; e < 8; ++e) {
                int m = rb + e;
                float sd = (float)sdB[m * 72 + n0s + colH];
                float nz = noise[(size_t)(g0 + m) * (Hh * Ss) + h * Ss + n0s + colH];
                float ns = (mean[e] + bdecv) + nz * sd;
                stateB[m * 72 + n0s + colH] = (_Float16)ns;
                xB[m * 104 + n0s + colH]    = (_Float16)ns;
            }
        }
        __syncthreads();

        // (g) reward MLP on (next_state, action) — waves 0-3
        if (w < 4) {
            v8f a1 = gemm_tile<3>(xB, 104, Wr1P, ws4, lane);
#pragma unroll
            for (int e = 0; e < 8; ++e) a1[e] = fmaxf(a1[e] + br1v, 0.0f);
            storeD(h1B, 72, n0s, lane, a1);
        }
        __syncthreads();
        if (w < 4) {
            v8f a2 = gemm_tile<2>(h1B, 72, Wr2P, ws4, lane);
#pragma unroll
            for (int e = 0; e < 8; ++e) a2[e] = fmaxf(a2[e] + br2v, 0.0f);
            storeD(h2B, 72, n0s, lane, a2);
        }
        __syncthreads();
        if (tid < ROWS) {   // one lane per row: final dot with Wr3 [64,1]
            float dot = br3v;
#pragma unroll
            for (int k = 0; k < Rh; ++k) dot += (float)h2B[tid * 72 + k] * Wr3[k];
            cumr += dot;
        }
        // next (a) barrier separates h2B readers from future writers
    }

    if (tid < ROWS) cumOut[g0 + tid] = cumr;
}

// ---------------------------------------------------------------------------
// argmax over C per batch row + gather best actions / value.
// One wave32 per batch element b; first-max tie-break matches jnp.argmax.
// ---------------------------------------------------------------------------
__global__ __launch_bounds__(256)
void select_kernel(const float* __restrict__ cum,      // [C*B] (c-major)
                   const float* __restrict__ actions,  // [C,B,H,A]
                   float* __restrict__ out) {          // [B*H*A] ++ [B]
    constexpr int HA = Hh * Aa;  // 160
    int wave = threadIdx.x >> 5, lane = threadIdx.x & 31;
    int b = blockIdx.x * 8 + wave;
    float v = (lane < Cc) ? cum[(size_t)lane * Bb + b] : -3.402823466e38f;
    int idx = lane;
#pragma unroll
    for (int off = 16; off; off >>= 1) {
        float ov = __shfl_xor(v, off, 32);
        int   oi = __shfl_xor(idx, off, 32);
        if (ov > v || (ov == v && oi < idx)) { v = ov; idx = oi; }
    }
    if (lane == 0) out[(size_t)Bb * HA + b] = v;   // best_value
    const float* src = actions + ((size_t)idx * Bb + b) * HA;
    for (int i = lane; i < HA; i += 32) out[(size_t)b * HA + i] = src[i];
}

extern "C" void kernel_launch(void* const* d_in, const int* in_sizes, int n_in,
                              void* d_out, int out_size, void* d_ws, size_t ws_size,
                              hipStream_t stream) {
    (void)in_sizes; (void)n_in; (void)out_size; (void)ws_size;
    const float* state0 = (const float*)d_in[0];
    const float* acts   = (const float*)d_in[1];
    const float* noi    = (const float*)d_in[2];
    const float* Wse  = (const float*)d_in[3];  const float* bse  = (const float*)d_in[4];
    const float* Wae  = (const float*)d_in[5];  const float* bae  = (const float*)d_in[6];
    const float* W1   = (const float*)d_in[7];  const float* b1   = (const float*)d_in[8];
    const float* W2   = (const float*)d_in[9];  const float* b2   = (const float*)d_in[10];
    const float* Wdec = (const float*)d_in[11]; const float* bdec = (const float*)d_in[12];
    const float* Wunc = (const float*)d_in[13]; const float* bunc = (const float*)d_in[14];
    const float* Wr1  = (const float*)d_in[15]; const float* br1  = (const float*)d_in[16];
    const float* Wr2  = (const float*)d_in[17]; const float* br2  = (const float*)d_in[18];
    const float* Wr3  = (const float*)d_in[19]; const float* br3  = (const float*)d_in[20];

    _Float16* wsP = (_Float16*)d_ws;
    float* cum = (float*)((char*)d_ws + CUM_BYTE_OFF);

    // pack f32 weights -> f16 WMMA B-fragment layout (deterministic each call)
    pack_w_kernel<<<32, 256, 0, stream>>>(Wse,  wsP + OFF_WSE, 64, 128, 64);
    pack_w_kernel<<<16, 256, 0, stream>>>(Wae,  wsP + OFF_WAE, 16, 128, 32);
    pack_w_kernel<<<128, 256, 0, stream>>>(W1,  wsP + OFF_W1, 256, 128, 256);
    pack_w_kernel<<<64, 256, 0, stream>>>(W2,   wsP + OFF_W2, 128, 128, 128);
    pack_w_kernel<<<32, 256, 0, stream>>>(Wdec, wsP + OFF_WDE, 128, 64, 128);
    pack_w_kernel<<<32, 256, 0, stream>>>(Wunc, wsP + OFF_WUN, 128, 64, 128);
    pack_w_kernel<<<24, 256, 0, stream>>>(Wr1,  wsP + OFF_WR1, 80, 64, 96);
    pack_w_kernel<<<16, 256, 0, stream>>>(Wr2,  wsP + OFF_WR2, 64, 64, 64);

    rollout_kernel<<<CB / ROWS, 256, 0, stream>>>(
        state0, acts, noi, wsP, bse, bae, b1, b2, bdec, bunc, br1, br2, br3, Wr3, cum);

    select_kernel<<<Bb / 8, 256, 0, stream>>>(cum, acts, (float*)d_out);
}
